// EchoStateNetwork_6116033429963
// MI455X (gfx1250) — matrix operand order repaired and checked by reading, not compiled, not run on hardware
//
#include <hip/hip_runtime.h>
#include <math.h>

// Echo State Network on MI455X (gfx1250, wave32).
//
// Strategy: single persistent kernel, 64 workgroups x 128 threads.
//  - W_res (2048x2048) converted to f16 and parked in LDS: 32 rows/WG x 2048 x 2B
//    = 128 KB of the WGP's 320 KB, loaded ONCE and reused for all 1024 steps.
//  - x pre-converted to f16 once (init kernel) so both GEMMs load A as b128.
//  - Per step each wave does 2 (input-proj) + 64 (recurrent) v_wmma_f32_16x16x32_f16,
//    fp32 accumulate, then tanh, then writes its 16x16 tile of h (f16) to a
//    ping-pong state buffer in d_ws.
//  - Device-wide sense-reversing barrier per step (atomics in d_ws + s_sleep spin):
//    1024 barriers instead of 1024 graph-node launches.
//  - Readout fused: after the barrier, WG w computes y[:, t, w] as f32 dot
//    products against its LDS-resident W_out row (deterministic, no atomics).
// Workspace: 256 B barrier + 2 x 32x2048 f16 state buffers + 32x1024x64 f16 x
//            = ~4.5 MB.

#define B_      32
#define T_      1024
#define NRES    2048
#define NIN     64
#define NOUT    64
#define NWG     64       // one 32-column chunk of the reservoir per workgroup
#define COLS    32
#define THREADS 128

typedef __attribute__((ext_vector_type(16))) _Float16 v16h;
typedef __attribute__((ext_vector_type(8)))  _Float16 v8h;
typedef __attribute__((ext_vector_type(8)))  float    v8f;

union AFrag { v16h v; v8h h[2]; };

__global__ void esn_init_kernel(unsigned* bar, unsigned* hbuf_u32,
                                const float* __restrict__ x,
                                _Float16* __restrict__ x16) {
  unsigned i = blockIdx.x * blockDim.x + threadIdx.x;
  unsigned stride = gridDim.x * blockDim.x;
  if (i == 0) { bar[0] = 0u; bar[1] = 0u; }
  // zero both ping-pong h buffers: 2 * 32 * 2048 halves = 65536 dwords
  for (unsigned j = i; j < (2u * B_ * NRES) / 2u; j += stride)
    hbuf_u32[j] = 0u;
  // pre-convert x [B,T,NIN] f32 -> f16 (2M elements, step-invariant)
  for (unsigned j = i; j < (unsigned)(B_ * T_ * NIN); j += stride)
    x16[j] = (_Float16)x[j];
}

__global__ __launch_bounds__(THREADS, 1)
void esn_persistent_kernel(const _Float16* __restrict__ x16,
                           const float* __restrict__ W_in,
                           const float* __restrict__ W_res,
                           const float* __restrict__ b_res,
                           const float* __restrict__ W_out,
                           const float* __restrict__ b_out,
                           float* __restrict__ y,
                           _Float16* __restrict__ hbuf,   // 2 * B_*NRES halves
                           unsigned* __restrict__ bar)
{
  __shared__ __attribute__((aligned(32))) _Float16 Wres_s[COLS * NRES]; // 128 KB
  __shared__ __attribute__((aligned(32))) _Float16 Win_s [COLS * NIN];  //   4 KB
  __shared__ float Wout_s[NRES];                                        //   8 KB
  __shared__ float red_s [THREADS];

  const int tid   = threadIdx.x;
  const int wg    = blockIdx.x;
  const int lane  = tid & 31;
  const int wid   = tid >> 5;
  const int mtile = wid >> 1;        // batch rows 16*mtile .. +15
  const int ntile = wid & 1;         // local cols 16*ntile .. +15

  // ---- one-time preload of this WG's weight slices into LDS ----
  for (int idx = tid; idx < COLS * NRES; idx += THREADS) {
    int i = idx >> 11, k = idx & (NRES - 1);
    Wres_s[idx] = (_Float16)W_res[(size_t)(wg * COLS + i) * NRES + k];
  }
  for (int idx = tid; idx < COLS * NIN; idx += THREADS) {
    int i = idx >> 6, d = idx & (NIN - 1);
    Win_s[idx] = (_Float16)W_in[(size_t)(wg * COLS + i) * NIN + d];
  }
  for (int k = tid; k < NRES; k += THREADS)
    Wout_s[k] = W_out[(size_t)wg * NRES + k];
  __syncthreads();

  // per-lane WMMA geometry (16-bit A: two 8-half runs at koff / 16+koff;
  // 16-bit B: same two 8-half runs at koff / 16+koff)
  const int am    = 16 * mtile + (lane & 15);      // batch row for A frag
  const int akoff = (lane >> 4) * 8;
  const int ncl   = 16 * ntile + (lane & 15);      // local column for B frag
  const int bkoff = (lane >> 4) * 16;
  const int cm0   = 16 * mtile + 8 * (lane >> 4);  // first batch row of C
  const int cng   = wg * COLS + ncl;               // global reservoir column
  const float rb  = b_res[cng];
  const float bo  = b_out[wg];

  const _Float16* __restrict__ pb_res = Wres_s + (size_t)ncl * NRES;
  const _Float16* __restrict__ pb_in  = Win_s  + (size_t)ncl * NIN;

  for (int t = 0; t < T_; ++t) {
    const _Float16* __restrict__ hprev = hbuf + (size_t)(t & 1)       * (B_ * NRES);
    _Float16* __restrict__       hcur  = hbuf + (size_t)((t + 1) & 1) * (B_ * NRES);

    v8f c = {};

    // fused input projection: c += x_t(16x64) @ W_in_chunk^T  (2 WMMAs)
    {
      const _Float16* __restrict__ pax = x16 + ((size_t)am * T_ + t) * NIN;
#pragma unroll
      for (int kb = 0; kb < NIN; kb += 32) {
        AFrag a;
        a.h[0] = *(const v8h*)(pax + kb + akoff);
        a.h[1] = *(const v8h*)(pax + kb + 16 + akoff);
        AFrag b;
        b.h[0] = *(const v8h*)(pb_in + kb + akoff);
        b.h[1] = *(const v8h*)(pb_in + kb + 16 + akoff);
        c = __builtin_amdgcn_wmma_f32_16x16x32_f16(false, a.v, false, b.v,
                                                   (short)0, c, false, false);
      }
    }

    // recurrent GEMM: c += h(16x2048) @ W_res_chunk^T  (64 WMMAs, B from LDS)
    {
      const _Float16* __restrict__ pah = hprev + (size_t)am * NRES;
#pragma unroll 4
      for (int kb = 0; kb < NRES; kb += 32) {
        AFrag a;
        a.h[0] = *(const v8h*)(pah + kb + akoff);
        a.h[1] = *(const v8h*)(pah + kb + 16 + akoff);
        AFrag b;
        b.h[0] = *(const v8h*)(pb_res + kb + akoff);
        b.h[1] = *(const v8h*)(pb_res + kb + 16 + akoff);
        c = __builtin_amdgcn_wmma_f32_16x16x32_f16(false, a.v, false, b.v,
                                                   (short)0, c, false, false);
      }
    }

    // h_new = tanh(c + b_res); store f16 tile (C layout: VGPR r -> M = cm0 + r)
#pragma unroll
    for (int r = 0; r < 8; ++r) {
      float hv = tanhf(c[r] + rb);
      hcur[(size_t)(cm0 + r) * NRES + cng] = (_Float16)hv;
    }

    // ---- device-wide barrier over all 64 workgroups ----
    __builtin_amdgcn_fence(__ATOMIC_RELEASE, "agent");  // flush own h stores
    __syncthreads();                                    // => whole WG flushed
    if (tid == 0) {
      unsigned g = __hip_atomic_load(&bar[1], __ATOMIC_RELAXED, __HIP_MEMORY_SCOPE_AGENT);
      unsigned a = __hip_atomic_fetch_add(&bar[0], 1u, __ATOMIC_ACQ_REL, __HIP_MEMORY_SCOPE_AGENT);
      if (a == NWG - 1) {
        __hip_atomic_store(&bar[0], 0u, __ATOMIC_RELAXED, __HIP_MEMORY_SCOPE_AGENT);
        __hip_atomic_fetch_add(&bar[1], 1u, __ATOMIC_RELEASE, __HIP_MEMORY_SCOPE_AGENT);
      } else {
        while (__hip_atomic_load(&bar[1], __ATOMIC_ACQUIRE, __HIP_MEMORY_SCOPE_AGENT) == g)
          __builtin_amdgcn_s_sleep(1);
      }
    }
    __syncthreads();
    __builtin_amdgcn_fence(__ATOMIC_ACQUIRE, "agent");  // invalidate for cross-WG h reads

    // ---- fused readout: y[b, t, wg] = b_out[wg] + h_t[b,:] . W_out[wg,:] ----
    {
      const int bb = tid & 31, sl = tid >> 5;            // 4 K-slices of 512
      const _Float16* __restrict__ ph = hcur + (size_t)bb * NRES + sl * 512;
      const float* __restrict__ pw = Wout_s + sl * 512;
      float acc = 0.f;
#pragma unroll 4
      for (int k0 = 0; k0 < 512; k0 += 8) {
        v8h hv = *(const v8h*)(ph + k0);
#pragma unroll
        for (int e = 0; e < 8; ++e) acc += (float)hv[e] * pw[k0 + e];
      }
      red_s[tid] = acc;
      __syncthreads();
      if (tid < 32) {
        float yv = bo + red_s[tid] + red_s[tid + 32] + red_s[tid + 64] + red_s[tid + 96];
        y[((size_t)tid * T_ + t) * NOUT + wg] = yv;
      }
      __syncthreads();
    }
  }
}

extern "C" void kernel_launch(void* const* d_in, const int* in_sizes, int n_in,
                              void* d_out, int out_size, void* d_ws, size_t ws_size,
                              hipStream_t stream) {
  const float* x     = (const float*)d_in[0];
  const float* W_in  = (const float*)d_in[1];
  const float* W_res = (const float*)d_in[2];
  const float* b_res = (const float*)d_in[3];
  const float* W_out = (const float*)d_in[4];
  const float* b_out = (const float*)d_in[5];
  float* y = (float*)d_out;

  unsigned* bar  = (unsigned*)d_ws;
  _Float16* hbuf = (_Float16*)((char*)d_ws + 256);            // 2 x 32x2048 f16
  _Float16* x16  = hbuf + (size_t)2 * B_ * NRES;              // 32x1024x64 f16

  esn_init_kernel<<<256, 256, 0, stream>>>(bar, (unsigned*)hbuf, x, x16);
  esn_persistent_kernel<<<NWG, THREADS, 0, stream>>>(x16, W_in, W_res, b_res,
                                                     W_out, b_out, y, hbuf, bar);
}
